// StructuredLinear_65189013619286
// MI455X (gfx1250) — matrix operand order repaired and checked by reading, not compiled, not run on hardware
//
#include <hip/hip_runtime.h>

typedef __attribute__((ext_vector_type(16))) _Float16 v16h;
typedef __attribute__((ext_vector_type(8)))  float    v8f;

#define N_DIM  2048
#define R_DIM  4
#define B_DIM  4096
#define N2     (2 * N_DIM)        // 4096: length of reversed/sign-baked tables

// ---------------------------------------------------------------------------
// Kernel 0: convert x (f32) -> f16, vectorized (4 elems / thread)
// ---------------------------------------------------------------------------
__global__ __launch_bounds__(256) void tplz_cvt_f16(const float* __restrict__ in,
                                                    _Float16* __restrict__ out) {
  size_t i = ((size_t)blockIdx.x * blockDim.x + threadIdx.x) * 4;
  float4 v = *(const float4*)(in + i);
  union { _Float16 h[4]; uint2 u; } t;
  t.h[0] = (_Float16)v.x; t.h[1] = (_Float16)v.y;
  t.h[2] = (_Float16)v.z; t.h[3] = (_Float16)v.w;
  *(uint2*)(out + i) = t.u;
}

// ---------------------------------------------------------------------------
// Kernel 1: M[p,q] = sum_i sum_k g_i[(p-k)%N] * h_i[(q-k)%N] * sgn(q>=k)
// K = R*N = 8192 WMMA GEMM; operands read from reversed, sign-baked LDS
// tables so fragment generation needs NO mod and NO sign select:
//   grl[j] = g[(2N-1-j) mod N]            -> A[p,kk] = grl[(N-1-p)+kk]
//   hrl[j] = h[(2N-1-j) mod N]*(j>=N?-1:1)-> B[kk,q] = hrl[(N-1-q)+kk]
// Each wave: 64x64 tile of M = 4x4 WMMA tiles. 8 waves/block, 128 blocks.
// ---------------------------------------------------------------------------
__global__ __launch_bounds__(256) void tplz_build_M(const float* __restrict__ G,
                                                    const float* __restrict__ H,
                                                    _Float16* __restrict__ M) {
  __shared__ _Float16 grl[R_DIM * N2];   // 32 KB
  __shared__ _Float16 hrl[R_DIM * N2];   // 32 KB
  const int tid = threadIdx.x;
  for (int idx = tid; idx < R_DIM * N2; idx += 256) {
    const int ci = idx >> 12;                      // / N2
    const int j  = idx & (N2 - 1);
    const int s  = (N2 - 1 - j) & (N_DIM - 1);     // (2N-1-j) mod N
    grl[idx] = (_Float16)G[ci * N_DIM + s];
    const float hv = H[ci * N_DIM + s];
    hrl[idx] = (_Float16)(j >= N_DIM ? -hv : hv);
  }
  __syncthreads();

  const int wave   = tid >> 5;
  const int lane   = tid & 31;
  const int lane16 = lane & 15;
  const int laneHi = lane >> 4;

  const int t     = blockIdx.x * 8 + wave;   // 0..1023
  const int pbase = (t >> 5) * 64;           // 32 tiles per dim
  const int qbase = (t & 31) * 64;

  v8f c[4][4] = {};

#pragma unroll
  for (int ci = 0; ci < R_DIM; ++ci) {
    // running base indices at kk0 = 0 (lane's laneHi K-offset baked in)
    int ja = (N_DIM - 1) - pbase - lane16 + laneHi * 16;   // A: grl index base
    int jb = (N_DIM - 1) - qbase - lane16 + laneHi * 16;   // B: hrl index base
    for (int kk0 = 0; kk0 < N_DIM; kk0 += 32) {
      v16h a[4], b[4];
#pragma unroll
      for (int rt = 0; rt < 4; ++rt)
#pragma unroll
        for (int e = 0; e < 16; ++e)   // A-frag K map: e (+16 for hi lanes, in ja)
          a[rt][e] = grl[ci * N2 + ja - rt * 16 + e];
#pragma unroll
      for (int ct = 0; ct < 4; ++ct)
#pragma unroll
        for (int e = 0; e < 16; ++e)   // B-frag K map: e (+16 for hi lanes, in jb)
          b[ct][e] = hrl[ci * N2 + jb - ct * 16 + e];
#pragma unroll
      for (int rt = 0; rt < 4; ++rt)
#pragma unroll
        for (int ct = 0; ct < 4; ++ct)
          c[rt][ct] = __builtin_amdgcn_wmma_f32_16x16x32_f16(
              false, a[rt], false, b[ct], (short)0, c[rt][ct], false, false);
      ja += 32;
      jb += 32;
    }
  }

#pragma unroll
  for (int rt = 0; rt < 4; ++rt)
#pragma unroll
    for (int ct = 0; ct < 4; ++ct)
#pragma unroll
      for (int v = 0; v < 8; ++v) {
        const int p = pbase + rt * 16 + v + laneHi * 8;
        const int q = qbase + ct * 16 + lane16;
        M[(size_t)p * N_DIM + q] = (_Float16)c[rt][ct][v];
      }
}

// ---------------------------------------------------------------------------
// Kernel 2: out[b,p] = sum_q X[b,q] * M[p,q]  (f16 operands, f32 accum)
// Each wave: 64x64 output tile = 4x4 WMMA tiles; fragments loaded straight
// from global as aligned 16B chunks (working set lives in the 192MB L2).
// Tiles: 64 (batch) x 32 (p) -> 2048 wave-tiles -> 256 blocks.
// ---------------------------------------------------------------------------
__global__ __launch_bounds__(256) void tplz_gemm_out(const _Float16* __restrict__ X,
                                                     const _Float16* __restrict__ M,
                                                     float* __restrict__ out) {
  const int tid    = threadIdx.x;
  const int wave   = tid >> 5;
  const int lane   = tid & 31;
  const int lane16 = lane & 15;
  const int laneHi = lane >> 4;

  const int t     = blockIdx.x * 8 + wave;   // 0..2047
  const int bbase = (t >> 5) * 64;           // 64 batch tiles
  const int pbase = (t & 31) * 64;           // 32 col tiles

  v8f c[4][4] = {};

  for (int qb = 0; qb < N_DIM; qb += 32) {
    v16h a[4], b[4];
    // A = X tile: row b; 16 contiguous halves at qb+laneHi*16
#pragma unroll
    for (int rt = 0; rt < 4; ++rt) {
      const _Float16* base =
          X + (size_t)(bbase + rt * 16 + lane16) * N_DIM + qb + laneHi * 16;
      union { uint4 u[2]; v16h v; } u;
      u.u[0] = *(const uint4*)base;
      u.u[1] = *(const uint4*)(base + 8);
      a[rt] = u.v;
    }
    // B = M^T tile: B[k=q, n=p] = M[p,q]; 16 contiguous halves at qb+laneHi*16
#pragma unroll
    for (int ct = 0; ct < 4; ++ct) {
      const _Float16* base =
          M + (size_t)(pbase + ct * 16 + lane16) * N_DIM + qb + laneHi * 16;
      union { uint4 u[2]; v16h v; } u;
      u.u[0] = *(const uint4*)base;
      u.u[1] = *(const uint4*)(base + 8);
      b[ct] = u.v;
    }
#pragma unroll
    for (int rt = 0; rt < 4; ++rt)
#pragma unroll
      for (int ct = 0; ct < 4; ++ct)
        c[rt][ct] = __builtin_amdgcn_wmma_f32_16x16x32_f16(
            false, a[rt], false, b[ct], (short)0, c[rt][ct], false, false);
  }

#pragma unroll
  for (int rt = 0; rt < 4; ++rt)
#pragma unroll
    for (int ct = 0; ct < 4; ++ct)
#pragma unroll
      for (int v = 0; v < 8; ++v) {
        const int bb = bbase + rt * 16 + v + laneHi * 8;
        const int pp = pbase + ct * 16 + lane16;
        out[(size_t)bb * N_DIM + pp] = c[rt][ct][v];
      }
}

// ---------------------------------------------------------------------------
extern "C" void kernel_launch(void* const* d_in, const int* in_sizes, int n_in,
                              void* d_out, int out_size, void* d_ws, size_t ws_size,
                              hipStream_t stream) {
  const float* x = (const float*)d_in[0];   // (B, N)
  const float* G = (const float*)d_in[1];   // (R, N)
  const float* H = (const float*)d_in[2];   // (R, N)
  float* out = (float*)d_out;               // (B, N)

  _Float16* Mh = (_Float16*)d_ws;                            // N*N f16   (8 MB)
  _Float16* Xh = (_Float16*)d_ws + (size_t)N_DIM * N_DIM;    // B*N f16  (16 MB)

  tplz_cvt_f16<<<(B_DIM * N_DIM) / 1024, 256, 0, stream>>>(x, Xh);
  tplz_build_M<<<128, 256, 0, stream>>>(G, H, Mh);
  tplz_gemm_out<<<256, 256, 0, stream>>>(Xh, Mh, out);
}
